// CascadeGNN_3942779978056
// MI455X (gfx1250) — compile-verified, matrix operand order checked by reading, not run on hardware
//
#include <hip/hip_runtime.h>
#include <hip/hip_bf16.h>

// ---------------------------------------------------------------------------
// CascadeGNN for MI455X (gfx1250, wave32, WMMA, LDS-staged weight panels)
// ---------------------------------------------------------------------------

#define NGB   256        // graphs per batch (B)
#define NPG   512        // nodes per data graph
#define NGN   (NGB*NPG)  // 131072 data nodes
#define EGN   (NGN*8)    // 1048576 data edges
#define NQPG  16
#define NQN   (NGB*NQPG) // 4096 query nodes
#define EQN   (NQN*8)    // 32768 query edges
#define FIN   64
#define FH    128
#define NL    2

typedef __bf16 bf16_t;
typedef __attribute__((ext_vector_type(16))) __bf16 v16bf;
typedef __attribute__((ext_vector_type(8)))  __bf16 v8bf;
typedef __attribute__((ext_vector_type(8)))  float  v8f;

// ------------------------- utility kernels ---------------------------------

__global__ void k_fill_u32(unsigned* p, unsigned v, size_t n) {
    size_t i = (size_t)blockIdx.x * blockDim.x + threadIdx.x;
    if (i < n) p[i] = v;
}

__global__ void k_cast_bf16(const float* __restrict__ in, bf16_t* __restrict__ out, size_t n) {
    size_t i = (size_t)blockIdx.x * blockDim.x + threadIdx.x;
    if (i < n) out[i] = (bf16_t)in[i];
}

// in: [K,N] f32 row-major  ->  out: [N,K] bf16 row-major (i.e. transposed)
__global__ void k_transpose_cast(const float* __restrict__ in, bf16_t* __restrict__ out,
                                 int K, int N) {
    size_t i = (size_t)blockIdx.x * blockDim.x + threadIdx.x;
    if (i >= (size_t)K * N) return;
    int k = (int)(i / N), n = (int)(i % N);
    out[(size_t)n * K + k] = (bf16_t)in[i];
}

// ------------------------- WMMA GEMM ---------------------------------------
// C[M,128] (+bias, optional relu) = A[M,K](bf16 row major) x Bt[128,K](= B^T).
// N is fixed at 128 for every GEMM in this network.
// Block = 256 threads = 8 waves. Each block: 128 M-rows x full 128 N-cols.
// The Bt panel (128 x K bf16, <= 64KB) is staged once into LDS with a
// 16B-granule XOR swizzle so fragment reads are bank-conflict free.
// Each wave holds 8 f32 accumulators (16x16 tiles spanning all 128 cols),
// so each A fragment fetched from global feeds 8 v_wmma ops; the B fragment
// stream is double-buffered so each ds_load overlaps the previous WMMA.

#define GEMM_N   128
#define GEMM_KMX 256

__device__ __forceinline__ v16bf pack_frag(v8bf lo, v8bf hi) {
    v16bf out;
#pragma unroll
    for (int i = 0; i < 8; ++i) { out[i] = lo[i]; out[8 + i] = hi[i]; }
    return out;
}

// CDNA5 16-bit A-fragment (16x32): lane l holds row (l&15); K groups
// { (l>>4)*8 .. +7 } and { 16+(l>>4)*8 .. +7 }.
__device__ __forceinline__ v16bf load_frag_g(const bf16_t* __restrict__ base,
                                             int ld, int row0, int k0) {
    int l    = threadIdx.x & 31;
    int half = l >> 4;
    const bf16_t* p = base + (size_t)(row0 + (l & 15)) * ld + k0 + half * 8;
    return pack_frag(*(const v8bf*)(p), *(const v8bf*)(p + 16));
}

__device__ __forceinline__ v16bf load_frag_lds(const bf16_t* Bs, int K, int mask,
                                               int row0, int k0) {
    int l  = threadIdx.x & 31;
    int n  = row0 + (l & 15);
    int g0 = (k0 >> 3) + (l >> 4);      // 16B granule index of low half
    int g1 = g0 + 2;                    // +16 elements
    const bf16_t* rowp = Bs + (size_t)n * K;
    v8bf lo = *(const v8bf*)(rowp + (((g0 ^ (n & mask)) << 3)));
    v8bf hi = *(const v8bf*)(rowp + (((g1 ^ (n & mask)) << 3)));
    return pack_frag(lo, hi);
}

__global__ void __launch_bounds__(256)
k_gemm_bf16(const bf16_t* __restrict__ A, const bf16_t* __restrict__ Bt,
            const float* __restrict__ bias,
            float* __restrict__ C, bf16_t* __restrict__ Cb,
            int M, int K, int relu) {
    __shared__ bf16_t Bs[GEMM_N * GEMM_KMX];   // 64 KB max

    const int mask    = (K >= 128) ? 15 : 7;   // granule XOR-swizzle mask
    const int gPerRow = K >> 3;

    // ---- stage Bt panel into LDS (swizzled), 16B per copy ----
    for (int gi = threadIdx.x; gi < GEMM_N * gPerRow; gi += 256) {
        int n = gi / gPerRow;
        int g = gi - n * gPerRow;
        int gp = g ^ (n & mask);
        *(v8bf*)(Bs + (size_t)n * K + (gp << 3)) =
            *(const v8bf*)(Bt + (size_t)n * K + (g << 3));
    }
    __syncthreads();

    int wave = threadIdx.x >> 5;
    int lane = threadIdx.x & 31;
    int m0   = blockIdx.x * 128 + wave * 16;

    v8f acc[8];
#pragma unroll
    for (int t = 0; t < 8; ++t) {
        float bn = bias ? bias[t * 16 + (lane & 15)] : 0.0f;
#pragma unroll
        for (int i = 0; i < 8; ++i) acc[t][i] = bn;
    }

    for (int k0 = 0; k0 < K; k0 += 32) {
        v16bf a = load_frag_g(A, K, m0, k0);
        v16bf b = load_frag_lds(Bs, K, mask, 0, k0);   // prime tile 0
#pragma unroll
        for (int t = 0; t < 8; ++t) {
            v16bf bnext = b;
            if (t < 7) bnext = load_frag_lds(Bs, K, mask, (t + 1) * 16, k0);
            acc[t] = __builtin_amdgcn_wmma_f32_16x16x32_bf16(
                         false, a, false, b, (short)0, acc[t], false, false);
            b = bnext;
        }
    }

    int mb = m0 + 8 * (lane >> 4);
#pragma unroll
    for (int v = 0; v < 8; ++v) {
        size_t rowoff = (size_t)(mb + v) * GEMM_N;
#pragma unroll
        for (int t = 0; t < 8; ++t) {
            int n = t * 16 + (lane & 15);
            float val = acc[t][v];
            if (relu) val = fmaxf(val, 0.0f);
            if (C)  C[rowoff + n]  = val;
            if (Cb) Cb[rowoff + n] = (bf16_t)val;
        }
    }
}

// ------------------------- AGNN kernels ------------------------------------

__device__ __forceinline__ unsigned enc_f(float f) {
    unsigned u = __float_as_uint(f);
    return (u & 0x80000000u) ? ~u : (u | 0x80000000u);
}
__device__ __forceinline__ float dec_f(unsigned u) {
    unsigned v = (u & 0x80000000u) ? (u & 0x7FFFFFFFu) : ~u;
    return __uint_as_float(v);
}

__global__ void k_inv_norm(const float* __restrict__ h, float* __restrict__ inv,
                           int n, int F) {
    int wid  = (int)(((size_t)blockIdx.x * blockDim.x + threadIdx.x) >> 5);
    int lane = threadIdx.x & 31;
    if (wid >= n) return;
    const float4* p = (const float4*)(h + (size_t)wid * F);
    float s = 0.0f;
    for (int c = lane; c < (F >> 2); c += 32) {
        float4 a = p[c];
        s += a.x * a.x + a.y * a.y + a.z * a.z + a.w * a.w;
    }
#pragma unroll
    for (int o = 16; o > 0; o >>= 1) s += __shfl_xor(s, o, 32);
    if (lane == 0) inv[wid] = rsqrtf(s + 1e-24f);
}

__global__ void k_edge_logit(const float* __restrict__ h, const float* __restrict__ inv,
                             const int* __restrict__ src, const int* __restrict__ dst,
                             float* __restrict__ e, unsigned* __restrict__ mEnc,
                             const float* __restrict__ betas, int layer,
                             int E, int F) {
    int eid  = (int)(((size_t)blockIdx.x * blockDim.x + threadIdx.x) >> 5);
    int lane = threadIdx.x & 31;
    if (eid >= E) return;
    int s = src[eid], d = dst[eid];
    const float4* ps = (const float4*)(h + (size_t)s * F);
    const float4* pd = (const float4*)(h + (size_t)d * F);
    float acc = 0.0f;
    for (int c = lane; c < (F >> 2); c += 32) {
        float4 a = ps[c], b = pd[c];
        acc += a.x * b.x + a.y * b.y + a.z * b.z + a.w * b.w;
    }
#pragma unroll
    for (int o = 16; o > 0; o >>= 1) acc += __shfl_xor(acc, o, 32);
    if (lane == 0) {
        float val = betas[layer] * acc * inv[s] * inv[d];
        e[eid] = val;
        atomicMax(&mEnc[d], enc_f(val));
    }
}

__global__ void k_edge_accum(const float* __restrict__ h, const float* __restrict__ e,
                             const unsigned* __restrict__ mEnc,
                             const int* __restrict__ src, const int* __restrict__ dst,
                             float* __restrict__ den, float* __restrict__ num,
                             int E, int F) {
    int eid  = (int)(((size_t)blockIdx.x * blockDim.x + threadIdx.x) >> 5);
    int lane = threadIdx.x & 31;
    if (eid >= E) return;
    int s = src[eid], d = dst[eid];
    float m = dec_f(mEnc[d]);
    if (!isfinite(m)) m = 0.0f;
    float ee = __expf(e[eid] - m);
    if (lane == 0) atomicAdd(&den[d], ee);
    const float4* ps = (const float4*)(h + (size_t)s * F);
    float* nd = num + (size_t)d * F;
    for (int c = lane; c < (F >> 2); c += 32) {
        float4 a = ps[c];
        float* np = nd + c * 4;
        atomicAdd(np + 0, ee * a.x);
        atomicAdd(np + 1, ee * a.y);
        atomicAdd(np + 2, ee * a.z);
        atomicAdd(np + 3, ee * a.w);
    }
}

// finalize: out = den>0 ? num/max(den,eps) : 0 ; optional f32 and bf16 outputs
__global__ void k_agnn_final(const float* __restrict__ num, const float* __restrict__ den,
                             float* __restrict__ outF, bf16_t* __restrict__ outB,
                             int n, int F) {
    size_t i = (size_t)blockIdx.x * blockDim.x + threadIdx.x;
    if (i >= (size_t)n * F) return;
    float d = den[i / F];
    float v = (d > 0.0f) ? num[i] / fmaxf(d, 1e-24f) : 0.0f;
    if (outF) outF[i] = v;
    if (outB) outB[i] = (bf16_t)v;
}

// ------------------------- pooling / concat / head -------------------------

// sum over contiguous nodesPerGraph rows; grid = B graphs, block = F threads
__global__ void k_pool_sum(const float* __restrict__ h, float* __restrict__ out,
                           int nodesPerGraph, int F) {
    int b = blockIdx.x, f = threadIdx.x;
    const float* p = h + (size_t)b * nodesPerGraph * F + f;
    float s = 0.0f;
    for (int i = 0; i < nodesPerGraph; ++i) {
        if (i + 8 < nodesPerGraph) __builtin_prefetch(p + (size_t)(i + 8) * F, 0, 0);
        s += p[(size_t)i * F];
    }
    out[(size_t)b * F + f] = s;
}

__global__ void k_concat(const float* __restrict__ hg, const float* __restrict__ hq_aggr,
                         const int* __restrict__ n2g, float* __restrict__ out,
                         int n, int F) {
    size_t i = (size_t)blockIdx.x * blockDim.x + threadIdx.x;
    if (i >= (size_t)n * 2 * F) return;
    int node = (int)(i / (2 * F));
    int c    = (int)(i % (2 * F));
    out[i] = (c < F) ? hg[(size_t)node * F + c]
                     : hq_aggr[(size_t)n2g[node] * F + (c - F)];
}

// y[b] = bp2 + dot(hp[b,:], Wp2[:,0]) ; one wave per graph
__global__ void k_head_out(const float* __restrict__ hp, const float* __restrict__ Wp2,
                           const float* __restrict__ bp2, float* __restrict__ y,
                           int nB, int F) {
    int wid  = (int)(((size_t)blockIdx.x * blockDim.x + threadIdx.x) >> 5);
    int lane = threadIdx.x & 31;
    if (wid >= nB) return;
    const float* p = hp + (size_t)wid * F;
    float s = 0.0f;
    for (int f = lane; f < F; f += 32) s += p[f] * Wp2[f];
#pragma unroll
    for (int o = 16; o > 0; o >>= 1) s += __shfl_xor(s, o, 32);
    if (lane == 0) y[wid] = s + bp2[0];
}

// ------------------------- host orchestration ------------------------------

extern "C" void kernel_launch(void* const* d_in, const int* in_sizes, int n_in,
                              void* d_out, int out_size, void* d_ws, size_t ws_size,
                              hipStream_t stream) {
    (void)in_sizes; (void)n_in; (void)out_size; (void)ws_size;

    const float* X       = (const float*)d_in[0];
    const float* Xq      = (const float*)d_in[1];
    const int*   g_src   = (const int*)d_in[2];
    const int*   g_dst   = (const int*)d_in[3];
    const int*   g_n2g   = (const int*)d_in[4];
    const int*   q_src   = (const int*)d_in[5];
    const int*   q_dst   = (const int*)d_in[6];
    const int*   q_n2g   = (const int*)d_in[7];  (void)q_n2g;
    const float* Wg      = (const float*)d_in[8];
    const float* bg      = (const float*)d_in[9];
    const float* Wq      = (const float*)d_in[10];
    const float* bq      = (const float*)d_in[11];
    const float* betas_g = (const float*)d_in[12];
    const float* betas_q = (const float*)d_in[13];
    const float* W1r     = (const float*)d_in[14];
    const float* b1r     = (const float*)d_in[15];
    const float* W2r     = (const float*)d_in[16];
    const float* b2r     = (const float*)d_in[17];
    const float* Wp1     = (const float*)d_in[18];
    const float* bp1     = (const float*)d_in[19];
    const float* Wp2     = (const float*)d_in[20];
    const float* bp2     = (const float*)d_in[21];
    float* y = (float*)d_out;

    // ---- workspace bump allocator ----
    char* wsp = (char*)d_ws;
    auto alloc = [&](size_t bytes) -> void* {
        void* p = wsp;
        wsp += (bytes + 255) & ~(size_t)255;
        return p;
    };

    bf16_t* Xbf     = (bf16_t*)alloc((size_t)NGN * FIN * 2);
    bf16_t* Xqbf    = (bf16_t*)alloc((size_t)NQN * FIN * 2);
    bf16_t* WgT     = (bf16_t*)alloc((size_t)FH * FIN * 2);
    bf16_t* WqT     = (bf16_t*)alloc((size_t)FH * FIN * 2);
    bf16_t* W1rT    = (bf16_t*)alloc((size_t)NL * FH * (2 * FH) * 2);
    bf16_t* W2rT    = (bf16_t*)alloc((size_t)NL * FH * FH * 2);
    bf16_t* Wp1T    = (bf16_t*)alloc((size_t)FH * FH * 2);
    float*  h_g     = (float*)alloc((size_t)NGN * FH * 4);
    float*  h_qA    = (float*)alloc((size_t)NQN * FH * 4);
    float*  h_qB    = (float*)alloc((size_t)NQN * FH * 4);
    float*  hq_aggr = (float*)alloc((size_t)NGB * FH * 4);
    float*  h_cat   = (float*)alloc((size_t)NGN * (2 * FH) * 4);
    float*  numg    = (float*)alloc((size_t)NGN * (2 * FH) * 4);
    bf16_t* hab     = (bf16_t*)alloc((size_t)NGN * (2 * FH) * 2);
    bf16_t* h_midB  = (bf16_t*)alloc((size_t)NGN * FH * 2);
    float*  invb    = (float*)alloc((size_t)NGN * 4);
    float*  ebuf    = (float*)alloc((size_t)EGN * 4);
    unsigned* mEnc  = (unsigned*)alloc((size_t)NGN * 4);
    float*  den     = (float*)alloc((size_t)NGN * 4);
    float*  hg_sum  = (float*)alloc((size_t)NGB * FH * 4);
    bf16_t* hgs_bf  = (bf16_t*)alloc((size_t)NGB * FH * 2);
    float*  hp      = (float*)alloc((size_t)NGB * FH * 4);

    auto blocks = [](size_t n, int t) { return (unsigned)((n + t - 1) / t); };
    const unsigned ENC_NEG_INF = 0x007FFFFFu;  // enc_f(-inf)

    auto fill = [&](void* p, unsigned v, size_t n) {
        k_fill_u32<<<blocks(n, 256), 256, 0, stream>>>((unsigned*)p, v, n);
    };
    auto cast = [&](const float* in, bf16_t* out, size_t n) {
        k_cast_bf16<<<blocks(n, 256), 256, 0, stream>>>(in, out, n);
    };
    auto transpose = [&](const float* in, bf16_t* out, int K, int N) {
        k_transpose_cast<<<blocks((size_t)K * N, 256), 256, 0, stream>>>(in, out, K, N);
    };
    auto gemm = [&](const bf16_t* A, const bf16_t* Bt, const float* bias,
                    float* C, bf16_t* Cb, int M, int K, int relu) {
        k_gemm_bf16<<<dim3(M / 128), 256, 0, stream>>>(A, Bt, bias, C, Cb, M, K, relu);
    };
    auto agnn = [&](const float* h, const int* src, const int* dst,
                    const float* betas, int layer, int nNodes, int E, int F,
                    float* num, float* outF, bf16_t* outB) {
        fill(mEnc, ENC_NEG_INF, (size_t)nNodes);
        fill(den, 0u, (size_t)nNodes);
        fill(num, 0u, (size_t)nNodes * F);
        k_inv_norm<<<blocks((size_t)nNodes * 32, 256), 256, 0, stream>>>(h, invb, nNodes, F);
        k_edge_logit<<<blocks((size_t)E * 32, 256), 256, 0, stream>>>(
            h, invb, src, dst, ebuf, mEnc, betas, layer, E, F);
        k_edge_accum<<<blocks((size_t)E * 32, 256), 256, 0, stream>>>(
            h, ebuf, mEnc, src, dst, den, num, E, F);
        k_agnn_final<<<blocks((size_t)nNodes * F, 256), 256, 0, stream>>>(
            num, den, outF, outB, nNodes, F);
    };

    // ---- weight prep (bf16 transposed) ----
    cast(X, Xbf, (size_t)NGN * FIN);
    cast(Xq, Xqbf, (size_t)NQN * FIN);
    transpose(Wg, WgT, FIN, FH);
    transpose(Wq, WqT, FIN, FH);
    for (int l = 0; l < NL; ++l) {
        transpose(W1r + (size_t)l * (2 * FH) * FH, W1rT + (size_t)l * FH * (2 * FH), 2 * FH, FH);
        transpose(W2r + (size_t)l * FH * FH,       W2rT + (size_t)l * FH * FH,       FH, FH);
    }
    transpose(Wp1, Wp1T, FH, FH);

    // ---- input projections ----
    gemm(Xbf,  WgT, bg, h_g,  nullptr, NGN, FIN, 0);
    gemm(Xqbf, WqT, bq, h_qA, nullptr, NQN, FIN, 0);

    float* hq_cur = h_qA;
    float* hq_nxt = h_qB;

    for (int l = 0; l < NL; ++l) {
        // query AGNN (F = H)
        agnn(hq_cur, q_src, q_dst, betas_q, l, NQN, EQN, FH,
             /*num=*/numg, /*outF=*/hq_nxt, /*outB=*/nullptr);
        { float* t = hq_cur; hq_cur = hq_nxt; hq_nxt = t; }

        // per-graph sum of query nodes
        k_pool_sum<<<NGB, FH, 0, stream>>>(hq_cur, hq_aggr, NQPG, FH);

        // concat [h_g | hq_aggr[g_n2g]]
        k_concat<<<blocks((size_t)NGN * 2 * FH, 256), 256, 0, stream>>>(
            h_g, hq_aggr, g_n2g, h_cat, NGN, FH);

        // data AGNN (F = 2H), bf16 output feeds MLP directly
        agnn(h_cat, g_src, g_dst, betas_g, l, NGN, EGN, 2 * FH,
             /*num=*/numg, /*outF=*/nullptr, /*outB=*/hab);

        // MLP reduce: relu(h @ W1r) @ W2r  -> new h_g
        gemm(hab, W1rT + (size_t)l * FH * (2 * FH), b1r + (size_t)l * FH,
             nullptr, h_midB, NGN, 2 * FH, 1);
        gemm(h_midB, W2rT + (size_t)l * FH * FH, b2r + (size_t)l * FH,
             h_g, nullptr, NGN, FH, 0);
    }

    // ---- readout head ----
    k_pool_sum<<<NGB, FH, 0, stream>>>(h_g, hg_sum, NPG, FH);
    cast(hg_sum, hgs_bf, (size_t)NGB * FH);
    gemm(hgs_bf, Wp1T, bp1, hp, nullptr, NGB, FH, 1);
    k_head_out<<<blocks((size_t)NGB * 32, 256), 256, 0, stream>>>(hp, Wp2, bp2, y, NGB, FH);
}
